// WeightOnlyInt8Linear_82291573391876
// MI455X (gfx1250) — matrix operand order repaired and checked by reading, not compiled, not run on hardware
//
#include <hip/hip_runtime.h>
#include <stdint.h>

typedef _Float16 v16h __attribute__((ext_vector_type(16)));
typedef _Float16 v8h  __attribute__((ext_vector_type(8)));
typedef _Float16 v2h  __attribute__((ext_vector_type(2)));
typedef float    v8f  __attribute__((ext_vector_type(8)));
typedef uint32_t v4u  __attribute__((ext_vector_type(4)));
typedef uint32_t v8u  __attribute__((ext_vector_type(8)));

#define TOKENS 128
#define IN_F   8192
#define OUT_F  28672
#define KB     64                  // K elements per pipelined stage
#define MTILE  128                 // all tokens in every block -> weights stream HBM exactly once
#define NTILE  128                 // cols per block (8 waves, 2 n-tiles x 4 m-tiles each)
#define XS_STRIDE (KB + 8)         // 72 halves = 36 dwords -> conflict-free b128 reads

// Dequant 2 packed int8 -> 2 fp16 in 3 full-rate VALU ops (exact):
//   v_perm_b32 -> {0,b_hi,0,b_lo}; ^0x64806480 -> fp16(1024+u); pk_add -1152 -> fp16(v)
__device__ __forceinline__ uint32_t dq2(uint32_t q, uint32_t sel) {
    uint32_t t = __builtin_amdgcn_perm(0u, q, sel) ^ 0x64806480u;
    v2h h = __builtin_bit_cast(v2h, t);
    const v2h off = {(_Float16)-1152.0f, (_Float16)-1152.0f};
    return __builtin_bit_cast(uint32_t, (v2h)(h + off));
}

__global__ __launch_bounds__(256)
void wo_int8_gemm(const _Float16* __restrict__ x,
                  const int8_t*   __restrict__ w,
                  const _Float16* __restrict__ scales,
                  const _Float16* __restrict__ bias,
                  _Float16*       __restrict__ out)
{
    __shared__ _Float16 Xs[2][MTILE * XS_STRIDE];   // 2 x 18 KB ping-pong buffers

    const int tid  = threadIdx.x;
    const int wave = tid >> 5;        // 0..7 (wave32)
    const int lane = tid & 31;
    const int l15  = lane & 15;
    const int sub  = lane >> 4;       // lane half: K 0-15 vs 16-31 of each 32-chunk

    const int nw = wave >> 1;         // 0..3 : n-position
    const int mw = wave & 1;          // 0..1 : m-position (4 m-tiles each)

    const int nbase = blockIdx.x * NTILE + nw * 32;   // 2 n-tiles per wave
    const int mloc  = mw * 64;                        // local m base within block

    const int8_t* wpn[2];
    wpn[0] = w + (size_t)(nbase + l15) * IN_F + sub * 16;
    wpn[1] = wpn[0] + (size_t)16 * IN_F;

    v8f acc[2][4] = {};   // [n-tile][m-tile] 16x16 f32 accumulators

    // ---- async stage of one 128 x 64 fp16 x-tile into LDS buffer (L2 -> LDS DMA path) ----
    auto stage_x = [&](int k0, int buf) {
        #pragma unroll
        for (int i = 0; i < 4; ++i) {
            int c   = tid + 256 * i;       // 128 rows * 8 chunks of 16B
            int row = c >> 3;
            int ch  = c & 7;
            const _Float16* gp = x + (size_t)row * IN_F + k0 + ch * 8;
            uint32_t ldsoff =
                (uint32_t)(uintptr_t)(&Xs[buf][row * XS_STRIDE + ch * 8]);
            asm volatile("global_load_async_to_lds_b128 %0, %1, off"
                         :: "v"(ldsoff), "v"(gp) : "memory");
        }
    };

    // ---- non-temporal weight prefetch: 2 n-tiles x 2 k-steps x 16B per lane ----
    auto load_w = [&](int k0, v4u (&wq)[2][2]) {
        #pragma unroll
        for (int nt = 0; nt < 2; ++nt)
            #pragma unroll
            for (int ks = 0; ks < 2; ++ks)
                wq[nt][ks] = __builtin_nontemporal_load(
                                 (const v4u*)(wpn[nt] + k0 + ks * 32));
    };

    // ---- one K=64 compute stage: 16 WMMAs fed from LDS A-tiles + register B-tiles ----
    auto compute = [&](const v4u (&wq)[2][2], int buf) {
        #pragma unroll
        for (int ks = 0; ks < 2; ++ks) {
            v16h a[4];
            #pragma unroll
            for (int mt = 0; mt < 4; ++mt) {
                const _Float16* p = &Xs[buf][(mloc + mt * 16 + l15) * XS_STRIDE
                                            + ks * 32 + sub * 16];
                v8h lo = *(const v8h*)(p);        // VGPRs 0-3 (K 0-7 / 16-23)
                v8h hi = *(const v8h*)(p + 8);    // VGPRs 4-7 (K 8-15 / 24-31)
                a[mt] = __builtin_shufflevector(lo, hi,
                        0,1,2,3,4,5,6,7,8,9,10,11,12,13,14,15);
            }
            v16h b[2];
            #pragma unroll
            for (int nt = 0; nt < 2; ++nt) {
                v4u q4 = wq[nt][ks];
                v8u bu;
                #pragma unroll
                for (int d = 0; d < 4; ++d) {
                    bu[2*d+0] = dq2(q4[d], 0x04010400u);  // {0,b1,0,b0}
                    bu[2*d+1] = dq2(q4[d], 0x04030402u);  // {0,b3,0,b2}
                }
                b[nt] = __builtin_bit_cast(v16h, bu);
            }
            #pragma unroll
            for (int mt = 0; mt < 4; ++mt)
                #pragma unroll
                for (int nt = 0; nt < 2; ++nt)
                    acc[nt][mt] = __builtin_amdgcn_wmma_f32_16x16x32_f16(
                        false, a[mt], false, b[nt],
                        (short)0, acc[nt][mt], false, false);
        }
    };

    // ---- software pipeline: prologue fills buffer 0 + wqA ----
    v4u wqA[2][2], wqB[2][2];
    load_w(0, wqA);
    stage_x(0, 0);
    asm volatile("s_wait_asynccnt 0x0" ::: "memory");
    __syncthreads();

    for (int k0 = 0; k0 < IN_F; k0 += 2 * KB) {
        // stage A: consume (wqA, buf0), prefetch (wqB, buf1) for k0+KB
        stage_x(k0 + KB, 1);
        load_w(k0 + KB, wqB);
        compute(wqA, 0);
        asm volatile("s_wait_asynccnt 0x0" ::: "memory");
        __syncthreads();

        // stage B: consume (wqB, buf1), prefetch (wqA, buf0) for k0+2KB
        if (k0 + 2 * KB < IN_F) {
            stage_x(k0 + 2 * KB, 0);
            load_w(k0 + 2 * KB, wqA);
        }
        compute(wqB, 1);
        asm volatile("s_wait_asynccnt 0x0" ::: "memory");
        __syncthreads();
    }

    // ---- epilogue: y = acc * scale[col] + bias[col], fp16 NT stores ----
    #pragma unroll
    for (int nt = 0; nt < 2; ++nt) {
        const int   col = nbase + nt * 16 + l15;
        const float sc  = (float)scales[col];
        const float bs  = (float)bias[col];
        #pragma unroll
        for (int mt = 0; mt < 4; ++mt)
            #pragma unroll
            for (int r = 0; r < 8; ++r) {
                int m = mloc + mt * 16 + r + 8 * sub;   // C/D layout: VGPR r -> M=r / M=8+r
                __builtin_nontemporal_store((_Float16)(acc[nt][mt][r] * sc + bs),
                                            &out[(size_t)m * OUT_F + col]);
            }
    }
}

extern "C" void kernel_launch(void* const* d_in, const int* in_sizes, int n_in,
                              void* d_out, int out_size, void* d_ws, size_t ws_size,
                              hipStream_t stream) {
    (void)in_sizes; (void)n_in; (void)d_ws; (void)ws_size; (void)out_size;
    const _Float16* x      = (const _Float16*)d_in[0];
    const int8_t*   w      = (const int8_t*)  d_in[1];
    const _Float16* scales = (const _Float16*)d_in[2];
    const _Float16* bias   = (const _Float16*)d_in[3];
    _Float16*       out    = (_Float16*)d_out;

    dim3 grid(OUT_F / NTILE);   // 224 blocks, each owns all 128 tokens x 128 cols
    dim3 block(256);            // 8 wave32 waves
    hipLaunchKernelGGL(wo_int8_gemm, grid, block, 0, stream, x, w, scales, bias, out);
}
